// MambaBlock_2027224564220
// MI455X (gfx1250) — matrix operand order repaired and checked by reading, not compiled, not run on hardware
//
#include <hip/hip_runtime.h>
#include <stdint.h>

// Problem constants (fixed by the reference setup)
#define B_   4
#define K_   4
#define R_   12
#define D_   32
#define N_   16
#define RD_  384      // d_inner = R_*D_
#define L_   4096     // H*W
#define CDIM 44       // R + 2N
#define CPAD 48       // padded to 3 WMMA column tiles
#define CH_  32       // scan chunks
#define CL_  128      // chunk length  (CH_*CL_ == L_)
#define NCHAIN (B_*K_*R_)   // 192 independent scan chains
#define PFD  16       // scan prefetch distance

typedef __attribute__((ext_vector_type(16))) __bf16 v16bf;
typedef __attribute__((ext_vector_type(8)))  float  v8f;

// ---------------------------------------------------------------------------
// Kernel 1: convert x_proj_weight (K,44,384) f32 -> (K,48,384) bf16, zero-pad
// ---------------------------------------------------------------------------
__global__ void prep_w(const float* __restrict__ W, __bf16* __restrict__ Wbf) {
  int i = blockIdx.x * blockDim.x + threadIdx.x;     // over K_*CPAD*RD_
  if (i >= K_ * CPAD * RD_) return;
  int d = i % RD_;
  int c = (i / RD_) % CPAD;
  int k = i / (RD_ * CPAD);
  float v = (c < CDIM) ? W[((size_t)k * CDIM + c) * RD_ + d] : 0.f;
  Wbf[i] = (__bf16)v;                                // hw v_cvt, RNE
}

// ---------------------------------------------------------------------------
// Kernel 2: fused CrossScan gather + bf16 WMMA projection + dt/dA postprocess.
// One wave per 16(l) x 48(c) strip; 3 f32 accumulators, 12 K-steps of 32.
// Outputs (all f32):
//   dt_ws,dA_ws : (B,K,R,L)   Bs_ws,Cs_ws : (B,K,N,L)
// ---------------------------------------------------------------------------
__global__ void __launch_bounds__(256)
proj_wmma(const float* __restrict__ x,
          const __bf16* __restrict__ Wbf,
          const float* __restrict__ A_logs,
          const float* __restrict__ dt_bias,
          const int* __restrict__ paths,
          float* __restrict__ dA_ws, float* __restrict__ dt_ws,
          float* __restrict__ Bs_ws, float* __restrict__ Cs_ws) {
  const int wave  = threadIdx.x >> 5;
  const int lane  = threadIdx.x & 31;
  const int strip = blockIdx.x * 8 + wave;      // 4096 strips total
  const int lt    = strip & 255;                // l-tile (16 rows of l)
  const int bk    = strip >> 8;
  const int b     = bk >> 2;
  const int k     = bk & 3;
  const int nn    = lane & 15;                  // A row m / B-C-D column n
  const int khalf = lane >> 4;
  const int kb    = khalf ? 16 : 0;             // A-fragment K base (contiguous, matches B)

  // gather index for this lane's l-row
  const int l_row = lt * 16 + nn;
  const int p     = paths[k * L_ + l_row];
  const float* xrow = x + (size_t)b * RD_ * L_ + p;   // xrow[d*L_] = x[b,d,p]

  v8f acc0 = {}; v8f acc1 = {}; v8f acc2 = {};
  const __bf16* wbase = Wbf + (size_t)k * CPAD * RD_;

  #pragma unroll 1
  for (int ds = 0; ds < RD_; ds += 32) {
    // A fragment: 16x32 bf16; lanes 0-15 K = 0..15, lanes 16-31 K = 16..31
    v16bf afrag;
    #pragma unroll
    for (int e = 0; e < 8; ++e) {
      afrag[e]     = (__bf16)xrow[(size_t)(ds + kb + e) * L_];
      afrag[e + 8] = (__bf16)xrow[(size_t)(ds + kb + 8 + e) * L_];
    }

    // B fragments: 32x16 bf16; lane column = nn, lane half selects K 0..15 / 16..31
    const __bf16* wk = wbase + (size_t)nn * RD_ + (ds + khalf * 16);
    v16bf b0 = *(const v16bf*)(wk);
    v16bf b1 = *(const v16bf*)(wk + 16 * RD_);
    v16bf b2 = *(const v16bf*)(wk + 32 * RD_);

    acc0 = __builtin_amdgcn_wmma_f32_16x16x32_bf16(false, afrag, false, b0, (short)0, acc0, false, false);
    acc1 = __builtin_amdgcn_wmma_f32_16x16x32_bf16(false, afrag, false, b1, (short)0, acc1, false, false);
    acc2 = __builtin_amdgcn_wmma_f32_16x16x32_bf16(false, afrag, false, b2, (short)0, acc2, false, false);
  }

  // C/D layout: lane column n = nn, rows m = v + 8*khalf  ->  l = lt*16 + khalf*8 + v
  v8f accs[3] = {acc0, acc1, acc2};
  const int lbase = lt * 16 + khalf * 8;
  #pragma unroll
  for (int t = 0; t < 3; ++t) {
    const int c = t * 16 + nn;
    v8f a = accs[t];
    if (c < R_) {                       // dt channel: softplus + dA = exp(dt*A)
      const float Ak   = -__expf(A_logs[k * R_ + c]);
      const float bias = dt_bias[k * R_ + c];
      const size_t base = ((size_t)((b * K_ + k) * R_ + c)) * L_ + lbase;
      #pragma unroll
      for (int v = 0; v < 8; ++v) {
        float z   = a[v] + bias;
        float dtv = (z > 20.f) ? z : __logf(1.f + __expf(z));   // softplus, hw trans
        dt_ws[base + v] = dtv;
        dA_ws[base + v] = __expf(dtv * Ak);
      }
    } else if (c < R_ + N_) {           // B channel
      const size_t base = ((size_t)((b * K_ + k) * N_ + (c - R_))) * L_ + lbase;
      #pragma unroll
      for (int v = 0; v < 8; ++v) Bs_ws[base + v] = a[v];
    } else if (c < CDIM) {              // C channel
      const size_t base = ((size_t)((b * K_ + k) * N_ + (c - R_ - N_))) * L_ + lbase;
      #pragma unroll
      for (int v = 0; v < 8; ++v) Cs_ws[base + v] = a[v];
    }                                    // c >= 44: padding, drop
  }
}

// ---------------------------------------------------------------------------
// Kernels 3/5: chunked selective scan. Diagonal transition dA is a SCALAR per
// (chain,l), so the chunk combine operator is (prodDA, localState).
// block = (chain,chunk), 512 threads = (d,n) state element.
// EMIT=0: h0=0, write S (chunk-local state) and P (prod dA).
// EMIT=1: h0=hstart, emit y (+ skip D*x) into ys (B,K,RD,L).
// ---------------------------------------------------------------------------
template <int EMIT>
__global__ void __launch_bounds__(512)
scan_pass(const float* __restrict__ x, const int* __restrict__ paths,
          const float* __restrict__ dA_ws, const float* __restrict__ dt_ws,
          const float* __restrict__ Bs_ws, const float* __restrict__ Cs_ws,
          const float* __restrict__ Ds, const float* __restrict__ hstart,
          float* __restrict__ S_ws, float* __restrict__ P_ws,
          float* __restrict__ ys_ws) {
  const int blk   = blockIdx.x;
  const int chain = blk / CH_;
  const int chunk = blk - chain * CH_;
  const int b  = chain / (K_ * R_);
  const int kr = chain - b * (K_ * R_);
  const int k  = kr / R_;
  const int r  = kr - k * R_;
  const int t  = threadIdx.x;
  const int d  = t >> 4;
  const int n  = t & 15;
  const int l0 = chunk * CL_;

  float h = EMIT ? hstart[(size_t)blk * 512 + t] : 0.f;
  const float Dval = Ds[(k * R_ + r) * D_ + d];

  const float* dAp = dA_ws + (size_t)chain * L_ + l0;
  const float* dtp = dt_ws + (size_t)chain * L_ + l0;
  const float* Bp  = Bs_ws + ((size_t)((b * K_ + k) * N_ + n)) * L_ + l0;
  const float* Cp  = Cs_ws + ((size_t)((b * K_ + k) * N_ + n)) * L_ + l0;
  const float* xp  = x + ((size_t)(b * RD_ + r * D_ + d)) * L_;   // index by p
  const int*   pp  = paths + k * L_ + l0;
  float*       yp  = ys_ws + ((size_t)((b * K_ + k) * RD_ + r * D_ + d)) * L_ + l0;

  float pacc = 1.f;
  for (int i = 0; i < CL_; ++i) {
    // data-dependent gather: prefetch ahead (global_prefetch_b8)
    if (i + PFD < CL_) __builtin_prefetch(&xp[pp[i + PFD]], 0, 3);
    const float da  = dAp[i];
    const float dtv = dtp[i];
    const float Bv  = Bp[i];
    const float xv  = xp[pp[i]];
    h = fmaf(da, h, dtv * xv * Bv);
    if (EMIT) {
      float yv = h * Cp[i];
      // reduce over n (16 lanes within half-wave; wave32)
      yv += __shfl_xor(yv, 1, 16);
      yv += __shfl_xor(yv, 2, 16);
      yv += __shfl_xor(yv, 4, 16);
      yv += __shfl_xor(yv, 8, 16);
      if (n == 0) yp[i] = fmaf(Dval, xv, yv);
    } else {
      pacc *= da;
    }
  }
  if (!EMIT) {
    S_ws[(size_t)blk * 512 + t] = h;
    if (t == 0) P_ws[blk] = pacc;
  }
}

// ---------------------------------------------------------------------------
// Kernel 4: prefix fix-up across chunks (32 sequential combine steps)
// ---------------------------------------------------------------------------
__global__ void __launch_bounds__(512)
scan_fix(const float* __restrict__ init, const float* __restrict__ S_ws,
         const float* __restrict__ P_ws, float* __restrict__ hstart) {
  const int chain = blockIdx.x;
  const int t  = threadIdx.x;
  const int kr = chain % (K_ * R_);               // init broadcast over batch
  float h = init[(size_t)kr * 512 + t];
  for (int c = 0; c < CH_; ++c) {
    const size_t idx = ((size_t)chain * CH_ + c) * 512 + t;
    hstart[idx] = h;
    h = fmaf(P_ws[chain * CH_ + c], h, S_ws[idx]);
  }
}

// ---------------------------------------------------------------------------
// Kernel 6: CrossMerge (inverse permutation per direction) + gate + sum over k
// ---------------------------------------------------------------------------
__global__ void merge_out(const float* __restrict__ ys, const float* __restrict__ gate,
                          const int* __restrict__ rev, float* __restrict__ out) {
  const int i = blockIdx.x * blockDim.x + threadIdx.x;
  if (i >= B_ * RD_ * L_) return;
  const int pidx = i & (L_ - 1);
  const int rd   = (i >> 12) % RD_;
  const int b    = i / (RD_ * L_);
  float acc = 0.f;
  #pragma unroll
  for (int k = 0; k < K_; ++k) {
    const int lr = rev[k * L_ + pidx];
    acc += ys[((size_t)((b * K_ + k) * RD_) + rd) * L_ + lr];
  }
  out[i] = acc * gate[i];
}

// ---------------------------------------------------------------------------
extern "C" void kernel_launch(void* const* d_in, const int* in_sizes, int n_in,
                              void* d_out, int out_size, void* d_ws, size_t ws_size,
                              hipStream_t stream) {
  const float* x      = (const float*)d_in[0];
  const float* gate   = (const float*)d_in[1];
  const float* W      = (const float*)d_in[2];
  const float* A_logs = (const float*)d_in[3];
  const float* dtb    = (const float*)d_in[4];
  const float* Ds     = (const float*)d_in[5];
  const float* init   = (const float*)d_in[6];
  const int*   paths  = (const int*)d_in[7];
  const int*   rev    = (const int*)d_in[8];
  float* out = (float*)d_out;

  // workspace carve-up (~134 MB total)
  char* ws = (char*)d_ws;
  size_t off = 0;
  auto carve = [&](size_t bytes) -> void* {
    void* p = ws + off;
    off = (off + bytes + 255) & ~(size_t)255;
    return p;
  };
  __bf16* Wbf  = (__bf16*)carve((size_t)K_ * CPAD * RD_ * 2);
  float* dA_ws = (float*)carve((size_t)B_ * K_ * R_ * L_ * 4);
  float* dt_ws = (float*)carve((size_t)B_ * K_ * R_ * L_ * 4);
  float* Bs_ws = (float*)carve((size_t)B_ * K_ * N_ * L_ * 4);
  float* Cs_ws = (float*)carve((size_t)B_ * K_ * N_ * L_ * 4);
  float* S_ws  = (float*)carve((size_t)NCHAIN * CH_ * 512 * 4);
  float* P_ws  = (float*)carve((size_t)NCHAIN * CH_ * 4);
  float* Hs_ws = (float*)carve((size_t)NCHAIN * CH_ * 512 * 4);
  float* ys_ws = (float*)carve((size_t)B_ * K_ * RD_ * L_ * 4);

  // 1. weights -> padded bf16
  prep_w<<<(K_ * CPAD * RD_ + 255) / 256, 256, 0, stream>>>(W, Wbf);
  // 2. gather + WMMA projection + dt/dA
  proj_wmma<<<512, 256, 0, stream>>>(x, Wbf, A_logs, dtb, paths,
                                     dA_ws, dt_ws, Bs_ws, Cs_ws);
  // 3. chunk-local scan (h0 = 0)
  scan_pass<0><<<NCHAIN * CH_, 512, 0, stream>>>(x, paths, dA_ws, dt_ws, Bs_ws, Cs_ws,
                                                 Ds, Hs_ws, S_ws, P_ws, ys_ws);
  // 4. cross-chunk prefix fix-up
  scan_fix<<<NCHAIN, 512, 0, stream>>>(init, S_ws, P_ws, Hs_ws);
  // 5. replay with correct chunk-start states, emit y
  scan_pass<1><<<NCHAIN * CH_, 512, 0, stream>>>(x, paths, dA_ws, dt_ws, Bs_ws, Cs_ws,
                                                 Ds, Hs_ws, S_ws, P_ws, ys_ws);
  // 6. cross-merge + gate
  merge_out<<<(B_ * RD_ * L_ + 255) / 256, 256, 0, stream>>>(ys_ws, gate, rev, out);
}
